// TtMoeLayer_70360154243135
// MI455X (gfx1250) — matrix operand order repaired and checked by reading, not compiled, not run on hardware
//
#include <hip/hip_runtime.h>
#include <hip/hip_bf16.h>

// ---------------- problem constants (from reference) ----------------
#define TOK   2048        // B*S tokens
#define HD    1024        // hidden H
#define FF    3584        // expert FFN F
#define NEXP  8           // experts
#define MT    32          // tokens per FFN block (2 WMMA M-tiles)
#define FQ    896         // F quarter held in LDS per pass (4 passes)

typedef __attribute__((ext_vector_type(8)))  float  v8f;
typedef __attribute__((ext_vector_type(16))) __bf16 v16bf;

union Frag {
  v16bf        v;
  __bf16       h[16];
  unsigned int ud[8];
};

// fast SiLU: x * sigmoid(x), sigmoid via single v_rcp_f32 (no exact-div sequence)
__device__ __forceinline__ float fast_silu(float x) {
  return x * __builtin_amdgcn_rcpf(1.0f + __expf(-x));
}

// ---------------- kernel 0: zero output + expert counters ----------------
__global__ void moe_zero_kernel(float* __restrict__ out, int* __restrict__ counts) {
  int i = blockIdx.x * 256 + threadIdx.x;
  if (i < TOK * HD) out[i] = 0.0f;
  if (i < NEXP)     counts[i] = 0;
}

// ---------------- kernel 1: gate logits -> top2 softmax routing + x->bf16 ----------------
__global__ __launch_bounds__(256) void moe_gate_kernel(
    const float* __restrict__ x, const float* __restrict__ gw,
    __bf16* __restrict__ xb, int* __restrict__ counts,
    int* __restrict__ tokIdx, float* __restrict__ tokW)
{
  const int t   = blockIdx.x;
  const int tid = threadIdx.x;
  const float* xr = x + (size_t)t * HD;

  float acc[NEXP];
#pragma unroll
  for (int e = 0; e < NEXP; ++e) acc[e] = 0.0f;

  for (int h = tid; h < HD; h += 256) {
    float xv = xr[h];
    xb[(size_t)t * HD + h] = (__bf16)xv;        // native cvt, read fp32 once
#pragma unroll
    for (int e = 0; e < NEXP; ++e) acc[e] += xv * gw[h * NEXP + e];
  }

  __shared__ float red[256][NEXP];
#pragma unroll
  for (int e = 0; e < NEXP; ++e) red[tid][e] = acc[e];
  __syncthreads();
  for (int s = 128; s > 0; s >>= 1) {
    if (tid < s) {
#pragma unroll
      for (int e = 0; e < NEXP; ++e) red[tid][e] += red[tid + s][e];
    }
    __syncthreads();
  }

  if (tid == 0) {
    float l0 = -3.4e38f; int i0 = 0;
#pragma unroll
    for (int e = 0; e < NEXP; ++e) { float v = red[0][e]; if (v > l0) { l0 = v; i0 = e; } }
    float l1 = -3.4e38f; int i1 = (i0 == 0) ? 1 : 0;
#pragma unroll
    for (int e = 0; e < NEXP; ++e) { if (e != i0) { float v = red[0][e]; if (v > l1) { l1 = v; i1 = e; } } }
    float e1  = __expf(l1 - l0);                // softmax over the selected 2 logits
    float inv = __builtin_amdgcn_rcpf(1.0f + e1);
    int s0 = atomicAdd(&counts[i0], 1);
    tokIdx[i0 * TOK + s0] = t; tokW[i0 * TOK + s0] = inv;
    int s1 = atomicAdd(&counts[i1], 1);
    tokIdx[i1 * TOK + s1] = t; tokW[i1 * TOK + s1] = e1 * inv;
  }
}

// ---- fragment loaders -------------------------------------------------
// A (16x32 bf16): lane half hf holds K = kb+16hf..+15 (matches B)
__device__ __forceinline__ Frag load_a_lds(const __bf16* row, int kb, int half) {
  Frag f;
  const unsigned int* p0 = (const unsigned int*)(row + kb + half * 16);
#pragma unroll
  for (int j = 0; j < 8; ++j) { f.ud[j] = p0[j]; }
  return f;
}
// B (32x16 bf16): lane half hf holds K = kb+16hf..+15, col = lane&15.
// fp32 source: 4x b128 loads + packed v_cvt_pk_bf16_f32 in registers.
__device__ __forceinline__ Frag load_b_f32(const float* base) {
  Frag f;
  const float4* p = (const float4*)base;
#pragma unroll
  for (int c = 0; c < 4; ++c) {
    float4 q = p[c];
    f.h[c * 4 + 0] = (__bf16)q.x; f.h[c * 4 + 1] = (__bf16)q.y;
    f.h[c * 4 + 2] = (__bf16)q.z; f.h[c * 4 + 3] = (__bf16)q.w;
  }
  return f;
}

// ---------------- kernel 2: per-expert SwiGLU FFN, 32-token tiles ----------------
__global__ __launch_bounds__(256) void moe_ffn_kernel(
    const __bf16* __restrict__ xb,
    const float* __restrict__ w1, const float* __restrict__ w3, const float* __restrict__ w2,
    const int* __restrict__ counts, const int* __restrict__ tokIdx, const float* __restrict__ tokW,
    float* __restrict__ out)
{
  const int e   = blockIdx.y;
  const int cnt = counts[e];
  const int m0  = blockIdx.x * MT;
  if (m0 >= cnt) return;                          // block-uniform early exit

  __shared__ __bf16 sX[MT * HD];                  // gathered x rows (64 KB)
  __shared__ __bf16 sAct[MT * FQ];                // activations, F quarter (56 KB)
  __shared__ int    sTok[MT];
  __shared__ float  sWt[MT];

  const int tid = threadIdx.x;
  if (tid < MT) {
    int g = m0 + tid;
    sTok[tid] = (g < cnt) ? tokIdx[e * TOK + g] : 0;
    sWt[tid]  = (g < cnt) ? tokW[e * TOK + g]  : 0.0f;  // padded rows contribute 0
  }
  __syncthreads();

  // async gather of 32 bf16 rows into LDS: global_load_async_to_lds_b128 (ASYNCcnt)
  for (int i = tid; i < MT * (HD / 8); i += 256) {      // 16-byte chunks
    int row = i >> 7, c = i & 127;
    const __bf16* g = xb + (size_t)sTok[row] * HD + (size_t)c * 8;
    unsigned int lds = (unsigned int)(unsigned long long)(sX + row * HD + c * 8);
    asm volatile("global_load_async_to_lds_b128 %0, %1, off"
                 :: "v"(lds), "v"(g) : "memory");
  }
  asm volatile("s_wait_asynccnt 0" ::: "memory");
  __syncthreads();

  const int lane = tid & 31;
  const int wave = tid >> 5;
  const int half = lane >> 4;
  const int ln   = lane & 15;                     // A-row (token) / B-col index

  const float* w1e = w1 + (size_t)e * FF * HD;
  const float* w3e = w3 + (size_t)e * FF * HD;
  const float* w2e = w2 + (size_t)e * HD * FF;

  v8f acc0[8], acc1[8];                           // out accumulators, live across quarters
#pragma unroll
  for (int j = 0; j < 8; ++j) { acc0[j] = (v8f){}; acc1[j] = (v8f){}; }

  for (int fq = 0; fq < FF / FQ; ++fq) {
    // ---- stage 1: act[MT, FQ] = silu(X W1^T) * (X W3^T); one B frag feeds 2 A tiles ----
    for (int ft = wave; ft < FQ / 16; ft += 8) {
      const int fl  = ft * 16;
      const int f0g = fq * FQ + fl;
      v8f ah0 = (v8f){}, ag0 = (v8f){}, ah1 = (v8f){}, ag1 = (v8f){};
      const float*  b1   = w1e + (size_t)(f0g + ln) * HD + half * 16;
      const float*  b3   = w3e + (size_t)(f0g + ln) * HD + half * 16;
      const __bf16* row0 = sX + ln * HD;
      const __bf16* row1 = sX + (16 + ln) * HD;
      for (int kt = 0; kt < HD / 32; ++kt) {
        const int kb = kt * 32;
        Frag fb1 = load_b_f32(b1 + kb);
        Frag fb3 = load_b_f32(b3 + kb);
        Frag a0  = load_a_lds(row0, kb, half);
        Frag a1  = load_a_lds(row1, kb, half);
        ah0 = __builtin_amdgcn_wmma_f32_16x16x32_bf16(false, a0.v, false, fb1.v, (short)0, ah0, false, false);
        ag0 = __builtin_amdgcn_wmma_f32_16x16x32_bf16(false, a0.v, false, fb3.v, (short)0, ag0, false, false);
        ah1 = __builtin_amdgcn_wmma_f32_16x16x32_bf16(false, a1.v, false, fb1.v, (short)0, ah1, false, false);
        ag1 = __builtin_amdgcn_wmma_f32_16x16x32_bf16(false, a1.v, false, fb3.v, (short)0, ag1, false, false);
      }
#pragma unroll
      for (int i = 0; i < 8; ++i) {               // D layout: M = i + 8*half, N = ln
        int M = i + half * 8;
        sAct[M        * FQ + fl + ln] = (__bf16)(fast_silu(ah0[i]) * ag0[i]);
        sAct[(16 + M) * FQ + fl + ln] = (__bf16)(fast_silu(ah1[i]) * ag1[i]);
      }
    }
    __syncthreads();

    // ---- stage 2 (partial): acc += act[MT, FQ] @ W2^T over this quarter ----
#pragma unroll
    for (int j = 0; j < 8; ++j) {
      const int h0 = (wave + j * 8) * 16;
      const float*  b2   = w2e + (size_t)(h0 + ln) * FF + fq * FQ + half * 16;
      const __bf16* row0 = sAct + ln * FQ;
      const __bf16* row1 = sAct + (16 + ln) * FQ;
      v8f a0c = acc0[j], a1c = acc1[j];
      for (int kt = 0; kt < FQ / 32; ++kt) {
        const int kb = kt * 32;
        Frag fb = load_b_f32(b2 + kb);
        Frag a0 = load_a_lds(row0, kb, half);
        Frag a1 = load_a_lds(row1, kb, half);
        a0c = __builtin_amdgcn_wmma_f32_16x16x32_bf16(false, a0.v, false, fb.v, (short)0, a0c, false, false);
        a1c = __builtin_amdgcn_wmma_f32_16x16x32_bf16(false, a1.v, false, fb.v, (short)0, a1c, false, false);
      }
      acc0[j] = a0c; acc1[j] = a1c;
    }
    __syncthreads();                              // before next quarter overwrites sAct
  }

  // ---- epilogue: scale by routing weight, scatter-add (each token in 2 experts) ----
#pragma unroll
  for (int j = 0; j < 8; ++j) {
    const int h0 = (wave + j * 8) * 16;
#pragma unroll
    for (int i = 0; i < 8; ++i) {
      const int M = i + half * 8;
      float v0 = acc0[j][i] * sWt[M];
      float v1 = acc1[j][i] * sWt[16 + M];
      __hip_atomic_fetch_add(&out[(size_t)sTok[M]      * HD + h0 + ln], v0,
                             __ATOMIC_RELAXED, __HIP_MEMORY_SCOPE_AGENT);
      __hip_atomic_fetch_add(&out[(size_t)sTok[16 + M] * HD + h0 + ln], v1,
                             __ATOMIC_RELAXED, __HIP_MEMORY_SCOPE_AGENT);
    }
  }
}

// ---------------- launch ----------------
extern "C" void kernel_launch(void* const* d_in, const int* in_sizes, int n_in,
                              void* d_out, int out_size, void* d_ws, size_t ws_size,
                              hipStream_t stream) {
  const float* x  = (const float*)d_in[0];   // [T, H]
  const float* gw = (const float*)d_in[1];   // [H, E]
  const float* w1 = (const float*)d_in[2];   // [E, F, H]
  const float* w3 = (const float*)d_in[3];   // [E, F, H]
  const float* w2 = (const float*)d_in[4];   // [E, H, F]
  float* out = (float*)d_out;                // [T, H]

  // workspace: bf16 x copy (4 MB), expert counters, per-expert token lists
  __bf16* xb     = (__bf16*)d_ws;
  int*    counts = (int*)((char*)d_ws + (size_t)TOK * HD * 2);
  int*    tokIdx = counts + NEXP;                                 // [E, T]
  float*  tokW   = (float*)(tokIdx + NEXP * TOK);                 // [E, T]

  moe_zero_kernel<<<(TOK * HD + 255) / 256, 256, 0, stream>>>(out, counts);
  moe_gate_kernel<<<TOK, 256, 0, stream>>>(x, gw, xb, counts, tokIdx, tokW);
  moe_ffn_kernel<<<dim3(TOK / MT, NEXP), 256, 0, stream>>>(xb, w1, w3, w2,
                                                           counts, tokIdx, tokW, out);
}
